// StackedSTU3D_48550310314363
// MI455X (gfx1250) — matrix operand order repaired and checked by reading, not compiled, not run on hardware
//
#include <hip/hip_runtime.h>
#include <hip/hip_bf16.h>
#include <math.h>

// ---------------------------------------------------------------------------
// StackedSTU3D for MI455X (gfx1250, wave32).
// irfftn(rfftn(z)*conj(Phi_f)) with the separable tied-index basis is a
// circular cross-correlation => three 1-D length-32 circular correlations
// (one wave32 line each, rotations via __shfl / ds_bpermute, basis rows
// regenerated from cos/sin). Channel contractions (Theta, fc1, fc2) are tall
// GEMMs (M=32768) on the matrix cores: v_wmma_f32_16x16x32_f16 with f32->f16
// converted A fragments and f16 weights pre-transposed to [K][N], staged
// global->LDS via the Tensor Data Mover (tensor_load_to_lds + tensorcnt).
// ---------------------------------------------------------------------------

#define NGRID   32
#define NPTS    32768           // 32^3
#define D       64
#define KBASIS  8
#define NLAYERS 4

typedef __attribute__((ext_vector_type(16))) _Float16 v16h;
typedef __attribute__((ext_vector_type(8)))  _Float16 v8h;
typedef __attribute__((ext_vector_type(8)))  float    v8f;

#if __has_builtin(__builtin_amdgcn_tensor_load_to_lds)
#define HAVE_TDM 1
#else
#define HAVE_TDM 0
#endif

#if HAVE_TDM
typedef __attribute__((ext_vector_type(4))) unsigned int u32x4;
typedef __attribute__((ext_vector_type(8))) int          i32x8;
typedef __attribute__((ext_vector_type(4))) int          i32x4;

// Stage a [kdim][ndim] f16 matrix from global into LDS with the TDM.
// D# per CDNA5 ISA: group0 = {count|flags, lds_addr,
// global_addr lo, global_addr hi | type=2}; group1 packs data_size=2B,
// tensor_dim0/1 = ndim/kdim, tile_dim0/1 = ndim/kdim, dim0 stride = ndim.
__device__ __forceinline__ void tdm_stage_weights(const _Float16* __restrict__ Wt,
                                                  _Float16* lds_base,
                                                  int kdim, int ndim)
{
    const unsigned long long ga = (unsigned long long)(uintptr_t)Wt;
    const unsigned lds_addr = (unsigned)(uintptr_t)lds_base;
    u32x4 g0;
    g0[0] = 1u;                                                 // count=1 (user D#)
    g0[1] = lds_addr;                                           // lds_addr
    g0[2] = (unsigned)(ga & 0xFFFFFFFFu);                       // global_addr[31:0]
    g0[3] = (unsigned)((ga >> 32) & 0x01FFFFFFu) | (2u << 30);  // [56:32] | type=2
    const unsigned s1 = (unsigned)(ndim * kdim);                // dim1 stride (elems)
    i32x8 g1;
    g1[0] = (int)(1u << 16);                                    // data_size=1 (2 bytes)
    g1[1] = (int)(((unsigned)ndim & 0xFFFFu) << 16);            // tensor_dim0[15:0]
    g1[2] = (int)((((unsigned)ndim >> 16) & 0xFFFFu) |
                  (((unsigned)kdim & 0xFFFFu) << 16));          // dim0 hi | dim1 lo
    g1[3] = (int)((((unsigned)kdim >> 16) & 0xFFFFu) |
                  (((unsigned)ndim & 0xFFFFu) << 16));          // dim1 hi | tile_dim0
    g1[4] = (int)((unsigned)kdim & 0xFFFFu);                    // tile_dim1, tile_dim2=0
    g1[5] = ndim;                                               // dim0_stride[31:0]
    g1[6] = (int)((s1 & 0xFFFFu) << 16);                        // dim1_stride[15:0]
    g1[7] = (int)(s1 >> 16);                                    // dim1_stride[47:16]
    i32x4 gz = (i32x4){0, 0, 0, 0};
#if __clang_major__ >= 23
    i32x8 gz8 = (i32x8){0, 0, 0, 0, 0, 0, 0, 0};
    __builtin_amdgcn_tensor_load_to_lds(g0, g1, gz, gz, gz8, 0);
#else
    __builtin_amdgcn_tensor_load_to_lds(g0, g1, gz, gz, 0);
#endif
}
#endif // HAVE_TDM

// ---------------------------------------------------------------------------
// Weight preparation: f32 [N][K] -> f16 [K][N] (transpose + convert), once
// per launch, so GEMM B-fragment rows are contiguous f16 in memory/LDS.
// ---------------------------------------------------------------------------
__global__ void wconv_kernel(const float* __restrict__ in, _Float16* __restrict__ out,
                             int nrows /*N*/, int kdim /*K*/)
{
    const int t = blockIdx.x * blockDim.x + threadIdx.x;
    if (t >= nrows * kdim) return;
    const int n  = t / kdim;
    const int kk = t - n * kdim;
    out[kk * nrows + n] = (_Float16)in[t];
}

// ---------------------------------------------------------------------------
// Lift: circular 3x3x3 conv, 6 in-channels (3 data + 3 coord) -> 64.
// ---------------------------------------------------------------------------
__global__ void lift_kernel(const float* __restrict__ x,
                            const float* __restrict__ w,
                            const float* __restrict__ bvec,
                            float* __restrict__ y, int b)
{
    __shared__ float staged[6 * 27];
    const int p  = blockIdx.x;
    const int xx = p >> 10, yy = (p >> 5) & 31, zz = p & 31;

    for (int t = threadIdx.x; t < 6 * 27; t += blockDim.x) {
        const int ci = t / 27, tap = t % 27;
        const int kx = tap / 9, ky = (tap / 3) % 3, kz = tap % 3;
        const int ix = (xx + kx + 31) & 31;
        const int iy = (yy + ky + 31) & 31;
        const int iz = (zz + kz + 31) & 31;
        float v;
        if (ci < 3) {
            v = x[(size_t)(b * 3 + ci) * NPTS + ix * 1024 + iy * 32 + iz];
        } else {
            const int g = (ci == 3) ? ix : ((ci == 4) ? iy : iz);
            v = (float)g * (1.0f / 31.0f);   // linspace(0,1,32)
        }
        staged[t] = v;
    }
    __syncthreads();

    const int o = threadIdx.x;               // output channel
    float acc = bvec[o];
    const float* wo = w + o * 162;           // w: (64, 6, 3,3,3) row-major
    #pragma unroll 6
    for (int t = 0; t < 162; ++t)
        acc = fmaf(staged[t], wo[t], acc);
    y[(size_t)o * NPTS + p] = acc;
}

// ---------------------------------------------------------------------------
// LayerNorm over channel dim at each point (coalesced over points).
// ---------------------------------------------------------------------------
__global__ void layernorm_kernel(const float* __restrict__ y,
                                 const float* __restrict__ gamma,
                                 const float* __restrict__ beta,
                                 float* __restrict__ z)
{
    const int p = blockIdx.x * blockDim.x + threadIdx.x;
    float s = 0.f, ss = 0.f;
    #pragma unroll 8
    for (int c = 0; c < D; ++c) {
        const float v = y[c * NPTS + p];
        s += v; ss = fmaf(v, v, ss);
    }
    const float mu  = s * (1.0f / D);
    const float var = ss * (1.0f / D) - mu * mu;
    const float rs  = rsqrtf(var + 1e-5f);
    #pragma unroll 8
    for (int c = 0; c < D; ++c) {
        const float v = y[c * NPTS + p];
        z[c * NPTS + p] = (v - mu) * rs * gamma[c] + beta[c];
    }
}

// ---------------------------------------------------------------------------
// 1-D circular correlation along axis of stride ST with Fourier-real basis
// row kk. One wave32 = one 32-point line; rotations via __shfl (ds_bpermute).
// ---------------------------------------------------------------------------
template <int ST>
__global__ void corr_kernel(const float* __restrict__ in,
                            float* __restrict__ out, int kk)
{
    const int lane = threadIdx.x & 31;
    const int wave = threadIdx.x >> 5;
    const int q    = blockIdx.x * (blockDim.x >> 5) + wave; // 0 .. 64*1024-1
    const int c    = q >> 10;
    const int rem  = q & 1023;

    int base;
    if      (ST == 1024) base = c * NPTS + rem;
    else if (ST == 32)   base = c * NPTS + (rem >> 5) * 1024 + (rem & 31);
    else                 base = c * NPTS + rem * 32;

    const float val = in[base + lane * ST];

    float vc;
    if (kk == 0) {
        vc = 0.17677669529663687f;                    // 1/sqrt(32)
    } else {
        const int   m   = (kk + 1) >> 1;              // harmonic index
        const float ang = 0.19634954084936207f * (float)(m * lane); // 2*pi/32
        vc = 0.25f * ((kk & 1) ? __cosf(ang) : __sinf(ang)); // sqrt(2/32)=0.25
    }

    float acc = 0.f;
    #pragma unroll 4
    for (int s = 0; s < 32; ++s) {
        const float vs = __shfl(vc, s, 32);
        const float zs = __shfl(val, (lane + s) & 31, 32);
        acc = fmaf(zs, vs, acc);
    }
    out[base + lane * ST] = acc;
}

// ---------------------------------------------------------------------------
// WMMA GEMM:  D[m,n] = A[m,:] . Wt[:,n]
//   A : f32 [KDIM][NPTS] channel-major (converted to f16 at fragment load)
//   Wt: f16 [KDIM][NDIM] (pre-transposed), staged global->LDS via TDM
// Each wave owns a 16 x NDIM strip; blockDim = 128 (4 waves).
// MODE 0: out += D                (Theta accumulation into S)
// MODE 1: out  = gelu(D + bias)   (fc1)
// MODE 2: out += D + bias         (fc2 + residual into y)
// ---------------------------------------------------------------------------
template <int MODE, int KDIM, int NDIM>
__global__ void gemm_wmma_kernel(const float* __restrict__ A,
                                 const _Float16* __restrict__ Wt,
                                 const float* __restrict__ bias,
                                 float* __restrict__ out)
{
    __shared__ __align__(32) _Float16 lds_w[KDIM * NDIM];

#if HAVE_TDM
    if (threadIdx.x < 32) {                      // wave 0 drives the TDM
        tdm_stage_weights(Wt, lds_w, KDIM, NDIM);
        __builtin_amdgcn_s_wait_tensorcnt(0);
    }
#else
    for (int t = threadIdx.x * 8; t < KDIM * NDIM; t += blockDim.x * 8)
        *(v8h*)&lds_w[t] = *(const v8h*)&Wt[t];
#endif
    __syncthreads();

    const int lane = threadIdx.x & 31;
    const int wave = threadIdx.x >> 5;
    const int m0   = (blockIdx.x * (blockDim.x >> 5) + wave) * 16;

    constexpr int NT = NDIM / 16;
    v8f acc[NT];
    #pragma unroll
    for (int t = 0; t < NT; ++t) acc[t] = (v8f){};

    // A-fragment lane mapping (16x32 f16, ISA 7.12.2):
    //   lanes 0-15 : row M=lane,    halves 0-7 -> K 0..7,  8-15 -> K 16..23
    //   lanes16-31 : row M=lane-16, halves 0-7 -> K 8..15, 8-15 -> K 24..31
    const int row  = m0 + (lane & 15);
    const int koff = (lane < 16) ? 0 : 8;
    // B-fragment (32x16 f16): lane holds column N=lane&15, K order as in A
    // (halves 0-7 -> K koff+j, 8-15 -> K 16+koff+j); K rows contiguous in lds_w.
    const int kb_lane = (lane & 15) + ((lane < 16) ? 0 : 16);

    for (int kb = 0; kb < KDIM; kb += 32) {
        v16h a;
        #pragma unroll
        for (int j = 0; j < 8; ++j) {
            a[j]     = (_Float16)A[(kb + koff + j)      * NPTS + row];
            a[8 + j] = (_Float16)A[(kb + 16 + koff + j) * NPTS + row];
        }
        if (kb + 32 < KDIM)
            __builtin_prefetch(&A[(kb + 32 + koff) * NPTS + row], 0, 1);

        const _Float16* wrow = &lds_w[(kb + koff) * NDIM + (lane & 15)];
        #pragma unroll
        for (int t = 0; t < NT; ++t) {
            v16h bf;
            #pragma unroll
            for (int j = 0; j < 8; ++j) {
                bf[j]     = wrow[j * NDIM + t * 16];
                bf[8 + j] = wrow[(16 + j) * NDIM + t * 16];
            }
            acc[t] = __builtin_amdgcn_wmma_f32_16x16x32_f16(
                /*neg_a=*/false, a, /*neg_b=*/false, bf,
                /*c_mod=*/(short)0, acc[t],
                /*reuse_a=*/false, /*reuse_b=*/false);
        }
    }

    // C/D layout: VGPR r -> row m0 + r (+8 for lanes 16-31), col = lane&15.
    const int ncol = lane & 15;
    const int mo   = (lane < 16) ? 0 : 8;
    #pragma unroll
    for (int t = 0; t < NT; ++t) {
        const int n  = t * 16 + ncol;
        const float bv = (MODE == 0) ? 0.f : bias[n];
        #pragma unroll
        for (int r = 0; r < 8; ++r) {
            float v = acc[t][r];
            const size_t idx = (size_t)n * NPTS + (m0 + mo + r);
            if (MODE == 0) {
                out[idx] += v;
            } else if (MODE == 1) {
                v += bv;
                out[idx] = 0.5f * v * (1.0f + erff(v * 0.70710678118654752f));
            } else {
                out[idx] += v + bv;
            }
        }
    }
}

// ---------------------------------------------------------------------------
// Final 1x1 projection d=64 -> Cout=3.
// ---------------------------------------------------------------------------
__global__ void proj_kernel(const float* __restrict__ y,
                            const float* __restrict__ pw,
                            const float* __restrict__ pb,
                            float* __restrict__ out)
{
    const int p = blockIdx.x * blockDim.x + threadIdx.x;
    float a0 = pb[0], a1 = pb[1], a2 = pb[2];
    #pragma unroll 8
    for (int c = 0; c < D; ++c) {
        const float v = y[c * NPTS + p];
        a0 = fmaf(pw[c],        v, a0);
        a1 = fmaf(pw[64 + c],   v, a1);
        a2 = fmaf(pw[128 + c],  v, a2);
    }
    out[p]            = a0;
    out[NPTS + p]     = a1;
    out[2 * NPTS + p] = a2;
}

// ---------------------------------------------------------------------------
extern "C" void kernel_launch(void* const* d_in, const int* in_sizes, int n_in,
                              void* d_out, int out_size, void* d_ws, size_t ws_size,
                              hipStream_t stream)
{
    (void)in_sizes; (void)n_in; (void)out_size; (void)ws_size;

    const float* x      = (const float*)d_in[0];   // (2,1,3,32,32,32)
    const float* lift_w = (const float*)d_in[1];   // (64,6,3,3,3)
    const float* lift_b = (const float*)d_in[2];   // (64)
    const float* gammas = (const float*)d_in[3];   // (4,64)
    const float* betas  = (const float*)d_in[4];   // (4,64)
    const float* Thetas = (const float*)d_in[5];   // (4,8,64,64)
    const float* fc1_w  = (const float*)d_in[6];   // (4,128,64)
    const float* fc1_b  = (const float*)d_in[7];   // (4,128)
    const float* fc2_w  = (const float*)d_in[8];   // (4,64,128)
    const float* fc2_b  = (const float*)d_in[9];   // (4,64)
    const float* proj_w = (const float*)d_in[10];  // (3,64)
    const float* proj_b = (const float*)d_in[11];  // (3)
    // d_in[12] = Phi_f: unused — basis regenerated analytically.

    float* ws = (float*)d_ws;
    float* Y  = ws;                       // 2*D*NPTS  residual stream
    float* Z  = Y  + 2 * D * NPTS;        // D*NPTS    layernorm output
    float* C1 = Z  + D * NPTS;            // D*NPTS    corr-x result
    float* C2 = C1 + D * NPTS;            // D*NPTS    corr-y result
    float* U  = C2 + D * NPTS;            // D*NPTS    corr-z result
    float* S  = U  + D * NPTS;            // D*NPTS    Theta accumulator
    float* H  = S  + D * NPTS;            // 2*D*NPTS  fc1 activation
    // f16 transposed weights after the f32 region (16B-aligned offsets).
    _Float16* th16 = (_Float16*)(H + 2 * D * NPTS);        // 4*8*64*64
    _Float16* f1t  = th16 + NLAYERS * KBASIS * 64 * 64;    // 4*64*128
    _Float16* f2t  = f1t  + NLAYERS * 64 * 128;            // 4*128*64

    // ---- one-time (per launch) weight transpose+convert to f16 [K][N] ----
    for (int l = 0; l < NLAYERS; ++l) {
        for (int kk = 0; kk < KBASIS; ++kk) {
            const size_t off = (size_t)(l * KBASIS + kk) * 64 * 64;
            wconv_kernel<<<16, 256, 0, stream>>>(Thetas + off, th16 + off, 64, 64);
        }
        wconv_kernel<<<32, 256, 0, stream>>>(fc1_w + (size_t)l * 128 * 64,
                                             f1t + (size_t)l * 64 * 128, 128, 64);
        wconv_kernel<<<32, 256, 0, stream>>>(fc2_w + (size_t)l * 64 * 128,
                                             f2t + (size_t)l * 128 * 64, 64, 128);
    }

    // ---- lift both batches ----
    for (int b = 0; b < 2; ++b)
        lift_kernel<<<NPTS, 64, 0, stream>>>(x, lift_w, lift_b,
                                             Y + (size_t)b * D * NPTS, b);

    // ---- stacked STU layers ----
    for (int l = 0; l < NLAYERS; ++l) {
        for (int b = 0; b < 2; ++b) {
            float* Yb = Y + (size_t)b * D * NPTS;

            layernorm_kernel<<<NPTS / 256, 256, 0, stream>>>(
                Yb, gammas + l * D, betas + l * D, Z);

            hipMemsetAsync(S, 0, (size_t)D * NPTS * sizeof(float), stream);

            for (int kk = 0; kk < KBASIS; ++kk) {
                corr_kernel<1024><<<8192, 256, 0, stream>>>(Z,  C1, kk);
                corr_kernel<32>  <<<8192, 256, 0, stream>>>(C1, C2, kk);
                corr_kernel<1>   <<<8192, 256, 0, stream>>>(C2, U,  kk);
                gemm_wmma_kernel<0, 64, 64><<<NPTS / 64, 128, 0, stream>>>(
                    U, th16 + (size_t)(l * KBASIS + kk) * 64 * 64, nullptr, S);
            }

            gemm_wmma_kernel<1, 64, 128><<<NPTS / 64, 128, 0, stream>>>(
                S, f1t + (size_t)l * 64 * 128, fc1_b + l * 128, H);
            gemm_wmma_kernel<2, 128, 64><<<NPTS / 64, 128, 0, stream>>>(
                H, f2t + (size_t)l * 128 * 64, fc2_b + l * 64, Yb);
        }
    }

    for (int b = 0; b < 2; ++b)
        proj_kernel<<<NPTS / 256, 256, 0, stream>>>(
            Y + (size_t)b * D * NPTS, proj_w, proj_b,
            (float*)d_out + (size_t)b * 3 * NPTS);
}
